// BoundaryNet_52046413693247
// MI455X (gfx1250) — compile-verified, matrix operand order checked by reading, not run on hardware
//
#include <hip/hip_runtime.h>
#include <stdint.h>

#define DD 128
#define TT 10
#define LL 4

typedef __attribute__((ext_vector_type(16))) __bf16 v16bf;
typedef __attribute__((ext_vector_type(8)))  float  v8f;

// ---------- helpers ----------
static __device__ __forceinline__ __bf16 f2bf(float f) {
  union { float f; uint32_t u; } a; a.f = f;
  uint32_t r = (a.u + 0x7FFFu + ((a.u >> 16) & 1u)) >> 16;
  union { uint16_t s; __bf16 b; } o; o.s = (uint16_t)r;
  return o.b;
}

union FragU { uint4 q[2]; v16bf v; };

// A fragment: 16x32 bf16 tile. ISA layout: lanes 0-15 hold M=lane, K groups
// {0..7,16..23}; lanes 16-31 hold M=lane-16, K groups {8..15,24..31}.
// `elem` is a 32-bit element offset so the backend can use SADDR+voffset.
static __device__ __forceinline__ v16bf load_a(const __bf16* __restrict__ X, uint32_t elem) {
  const __bf16* p = X + elem;
  FragU u;
  u.q[0] = *reinterpret_cast<const uint4*>(p);
  u.q[1] = *reinterpret_cast<const uint4*>(p + 16);
  return u.v;
}

// B fragment from LDS-staged pack: layout [jt][kc][lane][16 bf16]
static __device__ __forceinline__ v16bf load_b_lds(const __bf16* L, int jt, int KC,
                                                   int kc, int lane) {
  const __bf16* p = L + (uint32_t)(((jt * KC + kc) * 32 + lane) * 16);
  FragU u;
  u.q[0] = *reinterpret_cast<const uint4*>(p);
  u.q[1] = *reinterpret_cast<const uint4*>(p + 8);
  return u.v;
}

static __device__ __forceinline__ v8f wmma_bf16(v16bf a, v16bf b, v8f c) {
  return __builtin_amdgcn_wmma_f32_16x16x32_bf16(false, a, false, b, (short)0, c, false, false);
}

// ---------- weight pack: W[j][k] fp32 (row-major, K = KC*32) -> fragment-linear bf16 ----------
__global__ void k_pack(const float* __restrict__ W, __bf16* __restrict__ P, int KC) {
  int idx = blockIdx.x * blockDim.x + threadIdx.x;
  int total = 8 * KC * 512;                 // 8 jt * KC * 32 lanes * 16 elems
  if (idx >= total) return;
  int e    = idx & 15;
  int lane = (idx >> 4) & 31;
  int rem  = idx >> 9;
  int kc   = rem % KC;
  int jt   = rem / KC;
  int laneHi = lane >> 4, ln = lane & 15;
  int j = jt * 16 + ln;                     // B column n -> output feature j
  int k = kc * 32 + ((e < 8) ? ((laneHi ? 8 : 0) + e) : ((laneHi ? 24 : 16) + (e - 8)));
  P[idx] = f2bf(W[(uint32_t)(j * (KC * 32) + k)]);
}

// ---------- input-stage hidden layers (K=2, VALU is fine) ----------
__global__ void k_inhid(const float* __restrict__ ctrs, const float* __restrict__ feats,
                        const float* __restrict__ W1i, const float* __restrict__ b1i,
                        const float* __restrict__ W1s, const float* __restrict__ b1s,
                        __bf16* __restrict__ hin, __bf16* __restrict__ hseg, int Nn) {
  int idx = blockIdx.x * blockDim.x + threadIdx.x;
  if (idx >= Nn * DD) return;
  int n = idx >> 7, j = idx & 127;
  float x0 = ctrs[n * 2], x1 = ctrs[n * 2 + 1];
  hin[idx] = f2bf(fmaxf(W1i[j * 2] * x0 + W1i[j * 2 + 1] * x1 + b1i[j], 0.f));
  float y0 = feats[n * 2], y1 = feats[n * 2 + 1];
  hseg[idx] = f2bf(fmaxf(W1s[j * 2] * y0 + W1s[j * 2 + 1] * y1 + b1s[j], 0.f));
}

// ---------- generic WMMA GEMM: out[N,128] = act(X[N,KC*32] @ W^T + bias + resid) ----------
// Weight pack is staged block-wide into LDS; B fragments are preloaded 4-deep so
// ds_load waits are staggered and WMMAs issue back-to-back.
template <int KC>
__global__ void k_gemm(const __bf16* __restrict__ X, const __bf16* __restrict__ P,
                       const float* __restrict__ bias, const float* __restrict__ resid,
                       float* __restrict__ outF, __bf16* __restrict__ outB,
                       int Nrows, int relu) {
  extern __shared__ char smem_raw[];
  __bf16* lb = (__bf16*)smem_raw;
  {
    const uint4* src = (const uint4*)P;
    uint4* dst = (uint4*)lb;
    __builtin_prefetch(P, 0, 1);                    // global_prefetch_b8
#pragma unroll
    for (int i = 0; i < 2 * KC; i++)                // 512*KC uint4 total / 256 threads
      dst[threadIdx.x + 256 * i] = src[threadIdx.x + 256 * i];
  }
  __syncthreads();

  int lane = threadIdx.x & 31;
  int tile = blockIdx.x * (blockDim.x >> 5) + (threadIdx.x >> 5);
  int row0 = tile * 16;
  if (row0 >= Nrows) return;                        // wave-uniform, after barrier
  int laneHi = lane >> 4, ln = lane & 15;
  int arow = row0 + ln; if (arow >= Nrows) arow = Nrows - 1;

  v16bf a[KC];
#pragma unroll
  for (int kc = 0; kc < KC; kc++)
    a[kc] = load_a(X, (uint32_t)arow * (KC * 32) + kc * 32 + (laneHi ? 8 : 0));

#pragma unroll
  for (int jt = 0; jt < 8; jt++) {
    v8f acc = {};
#pragma unroll
    for (int kb = 0; kb < KC; kb += 4) {            // 4-deep B preload per chunk
      v16bf b[4];
#pragma unroll
      for (int u = 0; u < 4; u++) b[u] = load_b_lds(lb, jt, KC, kb + u, lane);
#pragma unroll
      for (int u = 0; u < 4; u++) acc = wmma_bf16(a[kb + u], b[u], acc);
    }
    int c = jt * 16 + ln;
#pragma unroll
    for (int r = 0; r < 8; r++) {
      int m = row0 + r + laneHi * 8;
      if (m < Nrows) {
        uint32_t o = ((uint32_t)m << 7) + (uint32_t)c;
        float v = acc[r];
        if (bias)  v += bias[c];
        if (resid) v += resid[o];
        if (relu)  v = fmaxf(v, 0.f);
        if (outF) outF[o] = v;
        if (outB) outB[o] = f2bf(v);
      }
    }
  }
}

// ---------- edge message: gather(feat[ev]) @ W^T, scatter-add to temp[eu] ----------
__global__ void k_edge(const __bf16* __restrict__ featb, const __bf16* __restrict__ P,
                       const int* __restrict__ eu, const int* __restrict__ ev,
                       float* __restrict__ temp, int Eedges) {
  extern __shared__ char smem_raw[];
  __bf16* lb = (__bf16*)smem_raw;
  {
    const uint4* src = (const uint4*)P;
    uint4* dst = (uint4*)lb;
    __builtin_prefetch(P, 0, 1);
#pragma unroll
    for (int i = 0; i < 8; i++)                     // 2048 uint4 / 256 threads
      dst[threadIdx.x + 256 * i] = src[threadIdx.x + 256 * i];
  }
  __syncthreads();

  int lane = threadIdx.x & 31;
  int tile = blockIdx.x * (blockDim.x >> 5) + (threadIdx.x >> 5);
  int e0 = tile * 16;
  if (e0 >= Eedges) return;                         // wave-uniform
  int laneHi = lane >> 4, ln = lane & 15;

  int src_row = ev[e0 + ln];                        // A row M = ln
  v16bf a[4];
#pragma unroll
  for (int kc = 0; kc < 4; kc++)
    a[kc] = load_a(featb, (uint32_t)src_row * DD + kc * 32 + (laneHi ? 8 : 0));

  uint32_t dofs[8];
#pragma unroll
  for (int r = 0; r < 8; r++)
    dofs[r] = ((uint32_t)eu[e0 + r + laneHi * 8]) << 7;

#pragma unroll
  for (int jt = 0; jt < 8; jt++) {
    v16bf b[4];
#pragma unroll
    for (int u = 0; u < 4; u++) b[u] = load_b_lds(lb, jt, 4, u, lane);
    v8f acc = {};
#pragma unroll
    for (int u = 0; u < 4; u++) acc = wmma_bf16(a[u], b[u], acc);
    uint32_t c = (uint32_t)(jt * 16 + ln);
#pragma unroll
    for (int r = 0; r < 8; r++)
      atomicAdd(temp + (dofs[r] + c), acc[r]);
  }
}

// ---------- GroupNorm(1, D) + affine + ReLU, one wave per node ----------
__global__ void k_gnorm(const float* __restrict__ temp, const float* __restrict__ g,
                        const float* __restrict__ b, __bf16* __restrict__ outB, int Nn) {
  int lane = threadIdx.x & 31;
  int node = blockIdx.x * (blockDim.x >> 5) + (threadIdx.x >> 5);
  if (node >= Nn) return;
  uint32_t base = (uint32_t)node << 7;
  float v[4], s = 0.f;
#pragma unroll
  for (int r = 0; r < 4; r++) { v[r] = temp[base + lane + 32 * r]; s += v[r]; }
#pragma unroll
  for (int off = 16; off; off >>= 1) s += __shfl_xor(s, off, 32);
  float mu = s * (1.f / 128.f), q = 0.f;
#pragma unroll
  for (int r = 0; r < 4; r++) { float d = v[r] - mu; q += d * d; }
#pragma unroll
  for (int off = 16; off; off >>= 1) q += __shfl_xor(q, off, 32);
  float inv = rsqrtf(q * (1.f / 128.f) + 1e-5f);
#pragma unroll
  for (int r = 0; r < 4; r++) {
    int c = lane + 32 * r;
    float y = (v[r] - mu) * inv * g[c] + b[c];
    outB[base + c] = f2bf(fmaxf(y, 0.f));
  }
}

// ---------- concat [feat_bf16 | embed[mark]] -> bf16 [N,256] ----------
__global__ void k_concat(const __bf16* __restrict__ featb, const float* __restrict__ embed,
                         const int* __restrict__ mark, __bf16* __restrict__ cat, int Nn) {
  int idx = blockIdx.x * blockDim.x + threadIdx.x;
  if (idx >= Nn * DD) return;
  int n = idx >> 7, j = idx & 127;
  cat[((uint32_t)n << 8) + j]      = featb[idx];
  cat[((uint32_t)n << 8) + DD + j] = f2bf(embed[(uint32_t)mark[n] * DD + j]);
}

// ---------- host ----------
extern "C" void kernel_launch(void* const* d_in, const int* in_sizes, int n_in,
                              void* d_out, int out_size, void* d_ws, size_t ws_size,
                              hipStream_t stream) {
  (void)n_in; (void)out_size; (void)ws_size;
  const int N = in_sizes[0] / 2;
  const int E = in_sizes[24] / TT;

  const float* ctrs   = (const float*)d_in[0];
  const float* featsI = (const float*)d_in[1];
  const float* in_W1  = (const float*)d_in[2];
  const float* in_b1  = (const float*)d_in[3];
  const float* in_W2  = (const float*)d_in[4];
  const float* in_b2  = (const float*)d_in[5];
  const float* seg_W1 = (const float*)d_in[6];
  const float* seg_b1 = (const float*)d_in[7];
  const float* seg_W2 = (const float*)d_in[8];
  const float* seg_b2 = (const float*)d_in[9];
  const float* embed  = (const float*)d_in[10];
  const float* metaW1 = (const float*)d_in[11];
  const float* metab1 = (const float*)d_in[12];
  const float* metaW2 = (const float*)d_in[13];
  const float* metab2 = (const float*)d_in[14];
  const float* ctrW   = (const float*)d_in[15];
  const float* edgeW  = (const float*)d_in[16];
  const float* normg  = (const float*)d_in[17];
  const float* normb  = (const float*)d_in[18];
  const float* c2W1   = (const float*)d_in[19];
  const float* c2b1   = (const float*)d_in[20];
  const float* c2W2   = (const float*)d_in[21];
  const float* c2b2   = (const float*)d_in[22];
  const int*   mark   = (const int*)d_in[23];
  const int*   edgeU  = (const int*)d_in[24];
  const int*   edgeV  = (const int*)d_in[25];
  float* out = (float*)d_out;

  // ws bump allocator
  char* wp = (char*)d_ws;
  auto alloc = [&](size_t bytes) -> void* {
    void* p = (void*)wp;
    wp += (bytes + 255) & ~(size_t)255;
    return p;
  };
  float*  temp  = (float*)alloc((size_t)N * DD * 4);   // accum / later reused as bf16 cat
  float*  feat  = (float*)alloc((size_t)N * DD * 4);
  __bf16* featb = (__bf16*)alloc((size_t)N * DD * 2);
  __bf16* h0b   = (__bf16*)alloc((size_t)N * DD * 2);
  __bf16* h1b   = (__bf16*)alloc((size_t)N * DD * 2);
  const size_t PK4 = 8 * 4 * 32 * 16;                  // elems for K=128 pack (32KB)
  const size_t PK8 = 8 * 8 * 32 * 16;                  // elems for K=256 pack (64KB)
  __bf16* pkInW2  = (__bf16*)alloc(PK4 * 2);
  __bf16* pkSegW2 = (__bf16*)alloc(PK4 * 2);
  __bf16* pkCtr   = (__bf16*)alloc(PK4 * 2 * LL);
  __bf16* pkEdge  = (__bf16*)alloc(PK4 * 2 * LL * TT);
  __bf16* pkC2a   = (__bf16*)alloc(PK4 * 2 * LL);
  __bf16* pkC2b   = (__bf16*)alloc(PK4 * 2 * LL);
  __bf16* pkM1    = (__bf16*)alloc(PK8 * 2);
  __bf16* pkM2    = (__bf16*)alloc(PK4 * 2);

  const int ntiles = (N + 15) / 16;
  const int gGemm  = (ntiles + 7) / 8;                 // 8 waves / block
  const int etiles = (E + 15) / 16;
  const int gEdge  = (etiles + 7) / 8;
  const int gElem  = (N * DD + 255) / 256;
  const int gNorm  = (N + 7) / 8;
  const int SH4 = 32768, SH8 = 65536;                  // LDS staging bytes

  // ---- pack all weights to bf16 fragment layout ----
  k_pack<<<64, 256, 0, stream>>>(in_W2, pkInW2, 4);
  k_pack<<<64, 256, 0, stream>>>(seg_W2, pkSegW2, 4);
  for (int i = 0; i < LL; i++) {
    k_pack<<<64, 256, 0, stream>>>(ctrW + (size_t)i * DD * DD, pkCtr + i * PK4, 4);
    k_pack<<<64, 256, 0, stream>>>(c2W1 + (size_t)i * DD * DD, pkC2a + i * PK4, 4);
    k_pack<<<64, 256, 0, stream>>>(c2W2 + (size_t)i * DD * DD, pkC2b + i * PK4, 4);
    for (int t = 0; t < TT; t++)
      k_pack<<<64, 256, 0, stream>>>(edgeW + ((size_t)(i * TT + t)) * DD * DD,
                                     pkEdge + (size_t)(i * TT + t) * PK4, 4);
  }
  k_pack<<<128, 256, 0, stream>>>(metaW1, pkM1, 8);
  k_pack<<<64, 256, 0, stream>>>(metaW2, pkM2, 4);

  // ---- input stage: feat = relu(mlp_in(ctrs) + mlp_seg(feats)) ----
  k_inhid<<<gElem, 256, 0, stream>>>(ctrs, featsI, in_W1, in_b1, seg_W1, seg_b1, h0b, h1b, N);
  k_gemm<4><<<gGemm, 256, SH4, stream>>>(h0b, pkInW2, in_b2, nullptr, temp, nullptr, N, 0);
  k_gemm<4><<<gGemm, 256, SH4, stream>>>(h1b, pkSegW2, seg_b2, temp, feat, featb, N, 1);

  // ---- fuse layers ----
  for (int i = 0; i < LL; i++) {
    k_gemm<4><<<gGemm, 256, SH4, stream>>>(featb, pkCtr + i * PK4, nullptr, nullptr,
                                           temp, nullptr, N, 0);
    for (int t = 0; t < TT; t++)
      k_edge<<<gEdge, 256, SH4, stream>>>(featb, pkEdge + (size_t)(i * TT + t) * PK4,
                                          edgeU + (size_t)t * E, edgeV + (size_t)t * E, temp, E);
    k_gnorm<<<gNorm, 256, 0, stream>>>(temp, normg + i * DD, normb + i * DD, h0b, N);
    k_gemm<4><<<gGemm, 256, SH4, stream>>>(h0b, pkC2a + i * PK4, c2b1 + i * DD, nullptr,
                                           nullptr, h1b, N, 1);
    // feat = relu(h1 @ W2^T + b2 + feat); residual read == write buffer (same element/thread)
    k_gemm<4><<<gGemm, 256, SH4, stream>>>(h1b, pkC2b + i * PK4, c2b2 + i * DD, feat,
                                           feat, featb, N, 1);
  }

  // ---- head: MLP(concat(feat, embed[mark])) ----
  __bf16* cat = (__bf16*)temp;               // reuse: N*256*2 == N*128*4 bytes
  k_concat<<<gElem, 256, 0, stream>>>(featb, embed, mark, cat, N);
  k_gemm<8><<<gGemm, 256, SH8, stream>>>(cat, pkM1, metab1, nullptr, nullptr, h0b, N, 1);
  k_gemm<4><<<gGemm, 256, SH4, stream>>>(h0b, pkM2, metab2, nullptr, out, nullptr, N, 0);
}
